// TransformerXL_37409165148670
// MI455X (gfx1250) — matrix operand order repaired and checked by reading, not compiled, not run on hardware
//
#include <hip/hip_runtime.h>
#include <hip/hip_bf16.h>

typedef unsigned short u16;
typedef unsigned int   u32;
typedef __attribute__((ext_vector_type(16))) __bf16 v16bf;
typedef __attribute__((ext_vector_type(8)))  float  v8f;

static constexpr int Bb = 4, Ss = 2048, DIMc = 1024, Hh = 16, Dd = 64;
static constexpr int WIN = 512;
static constexpr int ROWS = Bb * Ss;          // 8192
static constexpr int NQKV = 3 * Hh * Dd;      // 3072

// ---------- bf16 helpers ----------
__device__ inline u16 f2bf_bits(float x) {
  u32 u = __builtin_bit_cast(u32, x);
  u32 r = u + 0x7FFFu + ((u >> 16) & 1u);   // round-to-nearest-even
  return (u16)(r >> 16);
}
__device__ inline __bf16 f2bf(float x) {
  u16 b = f2bf_bits(x);
  __bf16 r; __builtin_memcpy(&r, &b, 2); return r;
}

// ---------- WMMA fragment loads (contiguous k = 16*(l>>4) + e, wave32) ----------
// A (16x32 bf16): lane l -> M = l&15 ; element e -> K = k0 + 16*(l>>4) + e
__device__ inline v16bf load_frag_a(const u16* __restrict__ A, int lda, int k0, int lane) {
  const u16* p = A + (size_t)(lane & 15) * lda + k0 + 16 * (lane >> 4);
  v16bf f;
  uint4* d = (uint4*)&f;
  d[0] = *(const uint4*)(p);
  d[1] = *(const uint4*)(p + 8);
  return f;
}
// B (32x16 bf16), Bt is N-major (Bt[n][k]): lane l -> N = l&15 ; K = k0 + 16*(l>>4) + e
__device__ inline v16bf load_frag_b(const u16* __restrict__ Bt, int ldb, int k0, int lane) {
  const u16* p = Bt + (size_t)(lane & 15) * ldb + k0 + 16 * (lane >> 4);
  v16bf f;
  uint4* d = (uint4*)&f;
  d[0] = *(const uint4*)(p);
  d[1] = *(const uint4*)(p + 8);
  return f;
}

#define WMMA_BF16(acc, a, b) \
  __builtin_amdgcn_wmma_f32_16x16x32_bf16(false, (a), false, (b), (short)0, (acc), false, false)

// async DMA: global -> LDS, 16 bytes per lane, tracked by ASYNCcnt
#define ASYNC_B128(ldsaddr, gaddr64) \
  asm volatile("global_load_async_to_lds_b128 %0, %1, off" \
               :: "v"(ldsaddr), "v"(gaddr64) : "memory")
#define WAIT_ASYNC(n) asm volatile("s_wait_asynccnt " #n ::: "memory")
#define WAIT_DS0()    asm volatile("s_wait_dscnt 0x0" ::: "memory")

// ---------- cast / pack kernels ----------
__global__ void cast_f32_to_bf16(const float* __restrict__ x, u16* __restrict__ y, long n) {
  long i = (long)blockIdx.x * blockDim.x + threadIdx.x;
  if (i < n) y[i] = f2bf_bits(x[i]);
}

// Wt[n][k] = concat(Wq, Wkv)[k][n]  (n in [0,3072), k in [0,1024))
__global__ void pack_wqkv_t(const float* __restrict__ Wq, const float* __restrict__ Wkv,
                            u16* __restrict__ Wt) {
  int k = blockIdx.x * blockDim.x + threadIdx.x;  // 0..1023
  int n = blockIdx.y;                             // 0..3071
  float v = (n < DIMc) ? Wq[(size_t)k * DIMc + n]
                       : Wkv[(size_t)k * (2 * DIMc) + (n - DIMc)];
  Wt[(size_t)n * DIMc + k] = f2bf_bits(v);
}

// Wt[n][k] = Wout[k][n]  (both 1024x1024)
__global__ void pack_wout_t(const float* __restrict__ Wout, u16* __restrict__ Wt) {
  int k = blockIdx.x * blockDim.x + threadIdx.x;
  int n = blockIdx.y;
  Wt[(size_t)n * DIMc + k] = f2bf_bits(Wout[(size_t)k * DIMc + n]);
}

// ---------- generic bf16 WMMA GEMM: C[M,N] = A[M,K] * Bt[N,K]^T, 64x64 per wave ----------
__global__ __launch_bounds__(32) void gemm_bf16_64x64(
    const u16* __restrict__ A, const u16* __restrict__ Bt, float* __restrict__ C,
    int M, int N, int K) {
  const int lane = threadIdx.x;
  const int m0 = blockIdx.y * 64;
  const int n0 = blockIdx.x * 64;
  v8f acc[4][4] = {};
  for (int k0 = 0; k0 < K; k0 += 32) {
    v16bf a[4], b[4];
#pragma unroll
    for (int i = 0; i < 4; ++i)
      a[i] = load_frag_a(A + (size_t)(m0 + 16 * i) * K, K, k0, lane);
#pragma unroll
    for (int j = 0; j < 4; ++j)
      b[j] = load_frag_b(Bt + (size_t)(n0 + 16 * j) * K, K, k0, lane);
    if (k0 + 32 < K) {  // pull next K-step toward L0 (global_prefetch_b8)
      __builtin_prefetch(A + (size_t)(m0 + (lane & 15)) * K + k0 + 32, 0, 0);
      __builtin_prefetch(Bt + (size_t)(n0 + (lane & 15)) * K + k0 + 32, 0, 0);
    }
#pragma unroll
    for (int i = 0; i < 4; ++i)
#pragma unroll
      for (int j = 0; j < 4; ++j)
        acc[i][j] = WMMA_BF16(acc[i][j], a[i], b[j]);
  }
  const int mo = 8 * (lane >> 4), nc = lane & 15;
#pragma unroll
  for (int i = 0; i < 4; ++i)
#pragma unroll
    for (int j = 0; j < 4; ++j)
#pragma unroll
      for (int r = 0; r < 8; ++r)
        C[(size_t)(m0 + 16 * i + r + mo) * N + (n0 + 16 * j + nc)] = acc[i][j][r];
}

// ---------- sigmoid(tokens @ Wmix) and sigmoid(tokens @ Wgate) : (B,H,S) each ----------
__global__ __launch_bounds__(32) void mixgate_kernel(
    const float* __restrict__ tokens, const float* __restrict__ Wmix,
    const float* __restrict__ Wgate, float* __restrict__ mixbuf, float* __restrict__ gatebuf) {
  int row = blockIdx.x;               // b*S + s
  int t = threadIdx.x;                // 32 threads: 16 heads x {gate, mix}
  int h = t & 15;
  int isMix = t >> 4;
  const float* W = isMix ? Wmix : Wgate;
  const float* x = tokens + (size_t)row * DIMc;
  float s = 0.f;
  for (int k = 0; k < DIMc; ++k) s = fmaf(x[k], W[k * Hh + h], s);
  float sg = 1.f / (1.f + __expf(-s));
  int b = row >> 11, n = row & (Ss - 1);
  float* dst = isMix ? mixbuf : gatebuf;
  dst[(size_t)(b * Hh + h) * Ss + n] = sg;
}

// ---------- RoPE(q,k) + value residual mix; emit bf16 q,k (b,h,s,d) and vT (b,h,d,s) ----------
__global__ void rope_mix_kernel(const float* __restrict__ qkv, const float* __restrict__ vres,
                                const float* __restrict__ mixbuf,
                                u16* __restrict__ qb, u16* __restrict__ kbuf,
                                u16* __restrict__ vT) {
  long idx = (long)blockIdx.x * blockDim.x + threadIdx.x;   // B*H*S
  if (idx >= (long)Bb * Hh * Ss) return;
  int s = (int)(idx & (Ss - 1));
  long bh = idx >> 11;
  int h = (int)(bh & (Hh - 1));
  int b = (int)(bh >> 4);
  const float* row  = qkv + (size_t)(b * Ss + s) * NQKV;
  const float* qrow = row + h * Dd;
  const float* krow = row + DIMc + h * Dd;
  const float* vrow = row + 2 * DIMc + h * Dd;
  const float* vr   = vres + ((size_t)bh * Ss + s) * Dd;
  float mix = mixbuf[(size_t)bh * Ss + s];
  const float scale = 0.125f;  // 64^-0.5
  u16* qd = qb   + ((size_t)bh * Ss + s) * Dd;
  u16* kd = kbuf + ((size_t)bh * Ss + s) * Dd;
  u16* vd = vT   + (size_t)bh * Dd * Ss + s;   // + d*Ss
#pragma unroll 4
  for (int i = 0; i < 32; ++i) {
    // inv_freq = 10000^(-2i/64) = exp(-i * 2*ln(10000)/64)
    float ang = (float)s * __expf(-(float)i * 0.2878231366242557f);
    float c = __cosf(ang), sn = __sinf(ang);
    float q0 = qrow[2 * i], q1 = qrow[2 * i + 1];
    float k0 = krow[2 * i], k1 = krow[2 * i + 1];
    qd[2 * i]     = f2bf_bits((q0 * c - q1 * sn) * scale);
    qd[2 * i + 1] = f2bf_bits((q1 * c + q0 * sn) * scale);
    kd[2 * i]     = f2bf_bits(k0 * c - k1 * sn);
    kd[2 * i + 1] = f2bf_bits(k1 * c + k0 * sn);
    float v0 = vrow[2 * i],     r0 = vr[2 * i];
    float v1 = vrow[2 * i + 1], r1 = vr[2 * i + 1];
    v0 = v0 + (r0 - v0) * mix;
    v1 = v1 + (r1 - v1) * mix;
    vd[(size_t)(2 * i) * Ss]     = f2bf_bits(v0);
    vd[(size_t)(2 * i + 1) * Ss] = f2bf_bits(v1);
  }
}

// ---------- windowed attention: one wave per 16 query rows of one (b,h) ----------
// K tiles and V chunks are staged in LDS via double-buffered async DMA (ASYNCcnt).
__global__ __launch_bounds__(32) void attn_window_kernel(
    const u16* __restrict__ qb, const u16* __restrict__ kbuf, const u16* __restrict__ vT,
    const float* __restrict__ gatebuf, u16* __restrict__ Ob) {
  __shared__ float sm[16][544];                      // score rows (window <= 528 cols)
  __shared__ __align__(16) u16 ktile[2][16][64];     // 2 x 2KB K tiles
  __shared__ __align__(16) u16 vtile[2][64][32];     // 2 x 4KB V chunks (d-major)
  const int lane = threadIdx.x;
  const int i0 = blockIdx.x * 16;
  const int h = blockIdx.y, b = blockIdx.z;
  const size_t bh = (size_t)(b * Hh + h);
  const u16* qbh = qb   + bh * Ss * Dd;
  const u16* kbh = kbuf + bh * Ss * Dd;
  const u16* vbh = vT   + bh * Dd * Ss;

  // q rows as A fragments (K = d, split 0..31 / 32..63), kept in registers
  v16bf qa0 = load_frag_a(qbh + (size_t)i0 * Dd, Dd, 0, lane);
  v16bf qa1 = load_frag_a(qbh + (size_t)i0 * Dd, Dd, 32, lane);

  int jstart = i0 - WIN; if (jstart < 0) jstart = 0;
  const int ntiles = (i0 + 16 - jstart) >> 4;   // <= 33
  const int ncols = ntiles * 16;
  const int nchunk = (ncols + 31) >> 5;
  const int mo = 8 * (lane >> 4), nc = lane & 15;

  auto issue_ktile = [&](int t, int buf) {
#pragma unroll
    for (int i = 0; i < 4; ++i) {
      int sidx = lane + 32 * i;                 // 0..127 ; row-of-16 x 8 segs
      int row = sidx >> 3, seg = sidx & 7;
      const u16* g = kbh + (size_t)(jstart + t * 16 + row) * Dd + seg * 8;
      u32 l = (u32)(uintptr_t)&ktile[buf][row][seg * 8];
      ASYNC_B128(l, (unsigned long long)(uintptr_t)g);
    }
  };
  auto issue_vchunk = [&](int c, int buf) {
#pragma unroll
    for (int i = 0; i < 8; ++i) {
      int sidx = lane + 32 * i;                 // 0..255 ; d-row x 4 segs
      int d = sidx >> 2, seg = sidx & 3;
      size_t joff = (size_t)(jstart + c * 32 + seg * 8);
      if (joff + 8 > (size_t)Ss) joff = 0;      // tail clamp: P is 0 there anyway
      const u16* g = vbh + (size_t)d * Ss + joff;
      u32 l = (u32)(uintptr_t)&vtile[buf][d][seg * 8];
      ASYNC_B128(l, (unsigned long long)(uintptr_t)g);
    }
  };

  // ---- pass 1: scores S = q k^T with window mask, K tiles via async LDS ----
  issue_ktile(0, 0);
  for (int t = 0; t < ntiles; ++t) {
    int buf = t & 1;
    if (t + 1 < ntiles) {
      WAIT_DS0();                 // reads of the buffer we are about to overwrite are done
      issue_ktile(t + 1, buf ^ 1);
      WAIT_ASYNC(0x4);            // tile t resident (only t+1's 4 DMAs outstanding)
    } else {
      WAIT_ASYNC(0x0);
    }
    const u16* kp = &ktile[buf][lane & 15][16 * (lane >> 4)];
    v16bf kb0, kb1;
    { uint4* dd = (uint4*)&kb0; dd[0] = *(const uint4*)kp;        dd[1] = *(const uint4*)(kp + 8); }
    { uint4* dd = (uint4*)&kb1; dd[0] = *(const uint4*)(kp + 32); dd[1] = *(const uint4*)(kp + 40); }
    v8f acc = {};
    acc = WMMA_BF16(acc, qa0, kb0);
    acc = WMMA_BF16(acc, qa1, kb1);
    int j0 = jstart + t * 16;
    int j = j0 + nc;
#pragma unroll
    for (int r = 0; r < 8; ++r) {
      int i = i0 + r + mo;
      int dist = i - j;
      sm[r + mo][j0 - jstart + nc] = (dist < 0 || dist > WIN) ? -1e30f : acc[r];
    }
  }
  // zero-fill pad cols [ncols, nchunk*32) so P@V chunks of 32 are safe
  for (int idx = lane; idx < 16 * (nchunk * 32 - ncols); idx += 32)
    sm[idx & 15][ncols + (idx >> 4)] = 0.f;

  // start V chunk 0 DMA now; it overlaps the softmax below
  issue_vchunk(0, 0);
  __syncthreads();

  // ---- softmax per row (16 rows handled by lanes 0..15) ----
  if (lane < 16) {
    float m = -1e30f;
    for (int c = 0; c < ncols; ++c) m = fmaxf(m, sm[lane][c]);
    float ssum = 0.f;
    for (int c = 0; c < ncols; ++c) { float e = __expf(sm[lane][c] - m); sm[lane][c] = e; ssum += e; }
    float inv = 1.f / ssum;
    for (int c = 0; c < ncols; ++c) sm[lane][c] *= inv;
  }
  __syncthreads();

  // ---- pass 2: O = P @ V, 32-key chunks, V via double-buffered async LDS ----
  v8f oacc[4] = {};
  for (int c = 0; c < nchunk; ++c) {
    int buf = c & 1;
    if (c + 1 < nchunk) {
      WAIT_DS0();
      issue_vchunk(c + 1, buf ^ 1);
      WAIT_ASYNC(0x8);            // chunk c resident (only c+1's 8 DMAs outstanding)
    } else {
      WAIT_ASYNC(0x0);
    }
    int kbase = c * 32;
    v16bf pa;
#pragma unroll
    for (int e = 0; e < 16; ++e) {
      int jl = kbase + e + 16 * (lane >> 4);  // A-layout K index
      pa[e] = f2bf(sm[lane & 15][jl]);
    }
#pragma unroll
    for (int dt = 0; dt < 4; ++dt) {
      const u16* vp = &vtile[buf][dt * 16 + (lane & 15)][16 * (lane >> 4)];
      v16bf vb;
      uint4* dd = (uint4*)&vb; dd[0] = *(const uint4*)vp; dd[1] = *(const uint4*)(vp + 8);
      oacc[dt] = WMMA_BF16(oacc[dt], pa, vb);
    }
  }

  // ---- epilogue: gate, store bf16 (b, s, h*64 + d) row-major for the out-proj GEMM ----
#pragma unroll
  for (int r = 0; r < 8; ++r) {
    int i = i0 + r + mo;
    float g = gatebuf[bh * Ss + i];
#pragma unroll
    for (int dt = 0; dt < 4; ++dt)
      Ob[(size_t)(b * Ss + i) * (Hh * Dd) + h * Dd + dt * 16 + nc] = f2bf_bits(oacc[dt][r] * g);
  }
}

// ---------------------------------------------------------------------------
extern "C" void kernel_launch(void* const* d_in, const int* in_sizes, int n_in,
                              void* d_out, int out_size, void* d_ws, size_t ws_size,
                              hipStream_t stream) {
  (void)in_sizes; (void)n_in; (void)out_size; (void)ws_size;
  const float* tokens = (const float*)d_in[0];
  const float* vres   = (const float*)d_in[1];
  const float* Wq     = (const float*)d_in[2];
  const float* Wkv    = (const float*)d_in[3];
  const float* Wout   = (const float*)d_in[4];
  const float* Wgate  = (const float*)d_in[5];
  const float* Wmix   = (const float*)d_in[6];
  float* out = (float*)d_out;

  // workspace layout (all 256B-aligned sizes)
  char* w = (char*)d_ws;
  u16*   tok_bf  = (u16*)w;   w += (size_t)ROWS * DIMc * 2;          // 16 MB
  u16*   wqkv_t  = (u16*)w;   w += (size_t)NQKV * DIMc * 2;          // 6 MB
  u16*   wout_t  = (u16*)w;   w += (size_t)DIMc * DIMc * 2;          // 2 MB
  float* qkv     = (float*)w; w += (size_t)ROWS * NQKV * 4;          // 96 MB
  float* mixbuf  = (float*)w; w += (size_t)Bb * Hh * Ss * 4;
  float* gatebuf = (float*)w; w += (size_t)Bb * Hh * Ss * 4;
  u16*   qbf     = (u16*)w;   w += (size_t)Bb * Hh * Ss * Dd * 2;    // 16 MB
  u16*   kbf     = (u16*)w;   w += (size_t)Bb * Hh * Ss * Dd * 2;
  u16*   vTbf    = (u16*)w;   w += (size_t)Bb * Hh * Dd * Ss * 2;
  u16*   Ob      = (u16*)w;   w += (size_t)ROWS * Hh * Dd * 2;

  // 1. casts / weight packing
  {
    long n = (long)ROWS * DIMc;
    cast_f32_to_bf16<<<dim3((n + 255) / 256), dim3(256), 0, stream>>>(tokens, tok_bf, n);
    pack_wqkv_t<<<dim3(DIMc / 256, NQKV), dim3(256), 0, stream>>>(Wq, Wkv, wqkv_t);
    pack_wout_t<<<dim3(DIMc / 256, DIMc), dim3(256), 0, stream>>>(Wout, wout_t);
  }
  // 2. QKV projection GEMM (8192 x 3072 x 1024), WMMA bf16
  gemm_bf16_64x64<<<dim3(NQKV / 64, ROWS / 64), dim3(32), 0, stream>>>(
      tok_bf, wqkv_t, qkv, ROWS, NQKV, DIMc);
  // 3. mix / gate sigmoids
  mixgate_kernel<<<dim3(ROWS), dim3(32), 0, stream>>>(tokens, Wmix, Wgate, mixbuf, gatebuf);
  // 4. RoPE + value-residual mix, emit bf16 operands for attention
  {
    long n = (long)Bb * Hh * Ss;
    rope_mix_kernel<<<dim3((n + 255) / 256), dim3(256), 0, stream>>>(
        qkv, vres, mixbuf, qbf, kbf, vTbf);
  }
  // 5. windowed attention (WMMA scores + WMMA P@V, async-LDS K/V staging)
  attn_window_kernel<<<dim3(Ss / 16, Hh, Bb), dim3(32), 0, stream>>>(
      qbf, kbf, vTbf, gatebuf, Ob);
  // 6. output projection GEMM (8192 x 1024 x 1024) -> f32 out
  gemm_bf16_64x64<<<dim3(DIMc / 64, ROWS / 64), dim3(32), 0, stream>>>(
      Ob, wout_t, out, ROWS, DIMc, DIMc);
}
